// NoExecution_4329327034941
// MI455X (gfx1250) — matrix operand order repaired and checked by reading, not compiled, not run on hardware
//
#include <hip/hip_runtime.h>
#include <hip/hip_bf16.h>
#include <math.h>

#define BB 32
#define TT 256
#define SS 32
#define HH 512
#define VV 128
#define EE 512

typedef __attribute__((ext_vector_type(16))) __bf16 v16bf;
typedef __attribute__((ext_vector_type(8)))  __bf16 v8bf;
typedef __attribute__((ext_vector_type(8)))  float  v8f;

// ---------------------------------------------------------------------------
// WMMA operand layouts (wave32, v_wmma_f32_16x16x32_bf16), contiguous k order used by this code:
// A (16x32 row-major): lane L (g=L/16, m=L%16): elems e -> K=kb+16g+e
//   (same order as B)  => two contiguous 16-byte bf16 loads.
// B from row-major W (B[k][n]=W[n][k]): lane L (n=L%16): elems e -> K=kb+16g+e
//   => one contiguous 32-byte bf16 load.
// D: VGPR v, lane L -> row v+8g, col L%16.
// ---------------------------------------------------------------------------
__device__ inline v16bf ld_a_bf(const __bf16* __restrict__ p, int g) {
  v8bf lo = *(const v8bf*)(p + 16 * g);
  v8bf hi = *(const v8bf*)(p + 16 * g + 8);
  return __builtin_shufflevector(lo, hi, 0, 1, 2, 3, 4, 5, 6, 7, 8, 9, 10, 11,
                                 12, 13, 14, 15);
}

__device__ inline v16bf ld_b_bf(const __bf16* __restrict__ p, int g) {
  return *(const v16bf*)(p + 16 * g);
}

__device__ inline v16bf zero16bf() {
  v16bf z;
#pragma unroll
  for (int e = 0; e < 16; ++e) z[e] = (__bf16)0.0f;
  return z;
}

__device__ inline v8f wmma_bf16(v16bf a, v16bf b, v8f c) {
  return __builtin_amdgcn_wmma_f32_16x16x32_bf16(false, a, false, b, (short)0, c,
                                                 false, false);
}

__device__ inline float sigmf(float x) { return 1.0f / (1.0f + expf(-x)); }

// Async global->LDS 16-byte copy (ASYNCcnt path) -----------------------------
__device__ inline void async_cp16(unsigned lds_off, const void* gaddr) {
  asm volatile("global_load_async_to_lds_b128 %0, %1, off"
               :: "v"(lds_off), "v"(gaddr)
               : "memory");
}
__device__ inline void async_wait0() {
  asm volatile("s_wait_asynccnt 0" ::: "memory");
}

// Sense-reversing grid barrier for persistent co-resident blocks -------------
__device__ inline void grid_barrier(unsigned* cnt, unsigned* gen, unsigned nblk) {
  __threadfence();
  if (threadIdx.x == 0) {
    volatile unsigned* vgen = gen;
    volatile unsigned* vcnt = cnt;
    unsigned g = *vgen;
    unsigned a = atomicAdd(cnt, 1u);
    if (a == nblk - 1u) {
      *vcnt = 0u;
      __threadfence();
      *vgen = g + 1u;
    } else {
      while (*vgen == g) __builtin_amdgcn_s_sleep(2);
    }
  }
  __threadfence();
}

// ---------------------------------------------------------------------------
__global__ void zero_out_kernel(float* out, int n) {
  int i = blockIdx.x * blockDim.x + threadIdx.x;
  if (i < n) out[i] = 0.0f;
}

__global__ void cvt_f32_bf16(const float* __restrict__ src,
                             __bf16* __restrict__ dst, long n) {
  long i = (long)blockIdx.x * blockDim.x + threadIdx.x;
  if (i < n) dst[i] = (__bf16)src[i];
}

// Y[M,N] = X[M,K] @ W[N,K]^T + bias; each wave computes 16x64 (4 N-tiles,
// A operand reused across 4 WMMAs -> 2.5 b128 loads per WMMA).
__global__ __launch_bounds__(256)
void gemm_bb4(const __bf16* __restrict__ X, const __bf16* __restrict__ W,
              const float* __restrict__ bias, float* __restrict__ Y,
              int M, int N, int K) {
  const int gw = (int)((blockIdx.x * blockDim.x + threadIdx.x) >> 5);
  const int lane = threadIdx.x & 31;
  const int nb = N >> 6;  // 64-col blocks
  const int total = (M >> 4) * nb;
  if (gw >= total) return;  // whole-wave exit, EXEC all-ones for WMMA
  const int mt = gw / nb, nt4 = gw - mt * nb;
  const int g = lane >> 4, r = lane & 15;
  const __bf16* arow = X + (long)(mt * 16 + r) * K;
  const __bf16* brow0 = W + (long)(nt4 * 64 + 0 * 16 + r) * K;
  const __bf16* brow1 = W + (long)(nt4 * 64 + 1 * 16 + r) * K;
  const __bf16* brow2 = W + (long)(nt4 * 64 + 2 * 16 + r) * K;
  const __bf16* brow3 = W + (long)(nt4 * 64 + 3 * 16 + r) * K;
  v8f a0 = {}, a1 = {}, a2 = {}, a3 = {};
  for (int k = 0; k < K; k += 32) {
    v16bf a = ld_a_bf(arow + k, g);
    a0 = wmma_bf16(a, ld_b_bf(brow0 + k, g), a0);
    a1 = wmma_bf16(a, ld_b_bf(brow1 + k, g), a1);
    a2 = wmma_bf16(a, ld_b_bf(brow2 + k, g), a2);
    a3 = wmma_bf16(a, ld_b_bf(brow3 + k, g), a3);
  }
#pragma unroll
  for (int c = 0; c < 4; ++c) {
    v8f acc = (c == 0) ? a0 : (c == 1) ? a1 : (c == 2) ? a2 : a3;
    const int n = nt4 * 64 + c * 16 + r;
    const float bv = bias ? bias[n] : 0.0f;
#pragma unroll
    for (int v = 0; v < 8; ++v)
      Y[(long)(mt * 16 + v + 8 * g) * N + n] = acc[v] + bv;
  }
}

// x_proj[B*T, 3H] = concat(emb[tokens], mask*scope[pid]) @ W_ih^T + b_ih
// (f32 out); 16x64 per wave.
__global__ __launch_bounds__(256)
void xproj_bf4(const __bf16* __restrict__ emb, const __bf16* __restrict__ scope,
               const int* __restrict__ tok, const int* __restrict__ pid,
               const unsigned char* __restrict__ pmask,
               const __bf16* __restrict__ W_ih, const float* __restrict__ b_ih,
               float* __restrict__ xproj) {
  const int N = 3 * HH, K = 2 * HH;
  const int gw = (int)((blockIdx.x * blockDim.x + threadIdx.x) >> 5);
  const int lane = threadIdx.x & 31;
  const int nb = N >> 6;  // 24
  const int mt = gw / nb, nt4 = gw - mt * nb;
  const int g = lane >> 4, r = lane & 15;
  const int m = mt * 16 + r;  // row in [0, B*T)
  const int b = m >> 8;       // T == 256
  const __bf16* row0 = emb + (long)tok[m] * HH;               // k in [0,H)
  const __bf16* row1 = scope + ((long)b * SS + pid[m]) * HH;  // k in [H,2H)
  const bool pm = pmask[m] != 0;
  const __bf16* brow0 = W_ih + (long)(nt4 * 64 + 0 * 16 + r) * K;
  const __bf16* brow1 = W_ih + (long)(nt4 * 64 + 1 * 16 + r) * K;
  const __bf16* brow2 = W_ih + (long)(nt4 * 64 + 2 * 16 + r) * K;
  const __bf16* brow3 = W_ih + (long)(nt4 * 64 + 3 * 16 + r) * K;
  const v16bf zz = zero16bf();
  v8f a0 = {}, a1 = {}, a2 = {}, a3 = {};
  for (int k = 0; k < K; k += 32) {  // k uniform -> uniform branch on half
    v16bf a;
    if (k < HH) {
      a = ld_a_bf(row0 + k, g);
    } else {
      a = ld_a_bf(row1 + (k - HH), g);
      a = pm ? a : zz;  // per-lane select, no divergence around WMMA
    }
    a0 = wmma_bf16(a, ld_b_bf(brow0 + k, g), a0);
    a1 = wmma_bf16(a, ld_b_bf(brow1 + k, g), a1);
    a2 = wmma_bf16(a, ld_b_bf(brow2 + k, g), a2);
    a3 = wmma_bf16(a, ld_b_bf(brow3 + k, g), a3);
  }
#pragma unroll
  for (int c = 0; c < 4; ++c) {
    v8f acc = (c == 0) ? a0 : (c == 1) ? a1 : (c == 2) ? a2 : a3;
    const int n = nt4 * 64 + c * 16 + r;
    const float bv = b_ih[n];
#pragma unroll
    for (int v = 0; v < 8; ++v)
      xproj[(long)(mt * 16 + v + 8 * g) * N + n] = acc[v] + bv;
  }
}

// Parallel persistent GRU: 64 single-wave blocks, one per (m-tile, 16-column
// j-tile). Each block async-stages its 48 KB W_hh slice (3 gates x 16 rows)
// into LDS ONCE; the 256-step recurrence then reads weights only from LDS.
// h state ping-pongs through two global bf16 buffers (native WMMA A format);
// each wave keeps its own 16x16 h tile in f32 registers for the z*h path.
// Steps are separated by a global sense-reversing barrier.
__global__ __launch_bounds__(32)
void gru_par(const float* __restrict__ xproj, const float* __restrict__ h0,
             const __bf16* __restrict__ W_hh, const float* __restrict__ b_hh,
             __bf16* __restrict__ o, __bf16* __restrict__ hping,
             __bf16* __restrict__ hpong, unsigned* cnt, unsigned* gen) {
  __shared__ __bf16 Wl[3 * 16 * HH];  // 48 KB
  const int blk = blockIdx.x;         // 64 blocks
  const int mt = blk >> 5;            // 0..1 (row tile)
  const int j = blk & 31;             // 0..31 (per-gate column tile)
  const int lane = threadIdx.x;
  const int g = lane >> 4, r = lane & 15;
  const int col = j * 16 + r;

  // One-time async stage of this block's weight slice into LDS.
  const unsigned ldsbase = (unsigned)(size_t)(&Wl[0]);  // low 32b = LDS offset
#pragma unroll 1
  for (int gate = 0; gate < 3; ++gate) {
    const char* src = (const char*)(W_hh + (long)(gate * HH + j * 16) * HH);
    for (int i = lane; i < (16 * HH * 2) / 16; i += 32)  // 1024 x 16B chunks
      async_cp16(ldsbase + (unsigned)(gate * 16 * HH * 2) + (unsigned)i * 16u,
                 src + (size_t)i * 16u);
  }

  // Init: my f32 h tile in registers + my bf16 tile of the ping buffer.
  float hold[8];
#pragma unroll
  for (int v = 0; v < 8; ++v) {
    const int row = mt * 16 + v + 8 * g;
    hold[v] = h0[row * HH + col];
    hping[row * HH + col] = (__bf16)hold[v];
  }
  async_wait0();
  grid_barrier(cnt, gen, 64u);  // all tiles of hping + LDS weights ready

  const float bhr = b_hh[col], bhz = b_hh[HH + col], bhn = b_hh[2 * HH + col];
  const __bf16* BrL = Wl + 0 * 16 * HH + r * HH;
  const __bf16* BzL = Wl + 1 * 16 * HH + r * HH;
  const __bf16* BnL = Wl + 2 * 16 * HH + r * HH;

  for (int t = 0; t < TT; ++t) {
    const __bf16* hsrc = (t & 1) ? hpong : hping;
    __bf16* hdst = (t & 1) ? hping : hpong;
    const __bf16* arow = hsrc + (mt * 16 + r) * HH;
    v8f aR = {}, aZ = {}, aN = {};
    for (int k = 0; k < HH; k += 32) {
      v16bf a = ld_a_bf(arow + k, g);  // global bf16 h, native layout
      aR = wmma_bf16(a, ld_b_bf(BrL + k, g), aR);  // LDS-resident weights
      aZ = wmma_bf16(a, ld_b_bf(BzL + k, g), aZ);
      aN = wmma_bf16(a, ld_b_bf(BnL + k, g), aN);
    }
#pragma unroll
    for (int v = 0; v < 8; ++v) {
      const int row = mt * 16 + v + 8 * g;  // batch index
      const long xb = ((long)row * TT + t) * (3 * HH) + col;
      const float xr = xproj[xb];
      const float xz = xproj[xb + HH];
      const float xn = xproj[xb + 2 * HH];
      const float rg = sigmf(xr + aR[v] + bhr);
      const float zg = sigmf(xz + aZ[v] + bhz);
      const float ng = tanhf(xn + rg * (aN[v] + bhn));
      const float hv = (1.0f - zg) * ng + zg * hold[v];
      hold[v] = hv;
      o[((long)row * TT + t) * HH + col] = (__bf16)hv;
      hdst[row * HH + col] = (__bf16)hv;
    }
    grid_barrier(cnt, gen, 64u);  // publish hdst before next step's reads
  }
}

// Token log-likelihood. bf16 W_out (128x512 = 128 KB) async-staged into LDS,
// reused by all 8 waves across 16 k-steps; A operand is the bf16 GRU output.
__global__ __launch_bounds__(256)
void tok_ll(const __bf16* __restrict__ o, const __bf16* __restrict__ W_out,
            const float* __restrict__ b_out, const int* __restrict__ tgt,
            float* __restrict__ out) {
  __shared__ __bf16 Wl[VV * HH];  // 128 KB
  __shared__ float lg[16 * VV];   // 8 KB
  const int tid = threadIdx.x;
  {
    const unsigned ldsbase = (unsigned)(size_t)(&Wl[0]);
    const char* gsrc = (const char*)W_out;
    for (int i = tid; i < (VV * HH * 2) / 16; i += 256)
      async_cp16(ldsbase + (unsigned)i * 16u, gsrc + (size_t)i * 16u);
    async_wait0();
  }
  __syncthreads();

  const int mt = blockIdx.x;  // 512 blocks, 16 rows each
  const int wave = tid >> 5, lane = tid & 31;
  const int g = lane >> 4, r = lane & 15;
  const int nt = wave;  // 8 waves x 16 cols = V
  const __bf16* arow = o + (long)(mt * 16 + r) * HH;
  const __bf16* brow = Wl + (nt * 16 + r) * HH;  // LDS-resident weights
  v8f acc = {};
  for (int k = 0; k < HH; k += 32)
    acc = wmma_bf16(ld_a_bf(arow + k, g), ld_b_bf(brow + k, g), acc);
  const float bv = b_out[nt * 16 + r];
#pragma unroll
  for (int v = 0; v < 8; ++v)
    lg[(v + 8 * g) * VV + nt * 16 + r] = acc[v] + bv;
  __syncthreads();
  if (tid < 16) {
    const int lr = tid;
    const int row = mt * 16 + lr;
    float mx = -1e30f;
    for (int jv = 0; jv < VV; ++jv) mx = fmaxf(mx, lg[lr * VV + jv]);
    float s = 0.0f;
    for (int jv = 0; jv < VV; ++jv) s += expf(lg[lr * VV + jv] - mx);
    const float ll = lg[lr * VV + tgt[row]] - mx - logf(s);
    atomicAdd(&out[row >> 8], ll);
  }
}

// Pointer attention LL: att[s] = sum_h tanh(hd + ok[s]) * v ; log_softmax(S=32)
__global__ __launch_bounds__(256)
void attn_ll(const float* __restrict__ hd, const float* __restrict__ okm,
             const float* __restrict__ vat, const int* __restrict__ tgt,
             const unsigned char* __restrict__ pmask, float* __restrict__ out) {
  __shared__ float red[256];
  __shared__ float att[SS];
  const int m = blockIdx.x;  // b*T + t
  const int tid = threadIdx.x;
  const int s = tid & 31, chunk = tid >> 5;  // 8 chunks x 64 h-elems
  const float* hr = hd + (long)m * HH;
  const float* orow = okm + ((long)(m >> 8) * SS + s) * HH;
  float p = 0.0f;
  const int hb = chunk * 64;
  for (int h = hb; h < hb + 64; ++h) p += tanhf(hr[h] + orow[h]) * vat[h];
  red[chunk * 32 + s] = p;
  __syncthreads();
  if (tid < SS) {
    float a = 0.0f;
    for (int c = 0; c < 8; ++c) a += red[c * 32 + tid];
    att[tid] = a;
  }
  __syncthreads();
  if (tid == 0 && pmask[m]) {
    float mx = -1e30f;
    for (int q = 0; q < SS; ++q) mx = fmaxf(mx, att[q]);
    float su = 0.0f;
    for (int q = 0; q < SS; ++q) su += expf(att[q] - mx);
    const float ll = att[tgt[m]] - mx - logf(su);
    atomicAdd(&out[m >> 8], ll);
  }
}

// ---------------------------------------------------------------------------
static inline void launch_cvt(const float* s, __bf16* d, long n, hipStream_t st) {
  int blocks = (int)((n + 255) / 256);
  cvt_f32_bf16<<<blocks, 256, 0, st>>>(s, d, n);
}

extern "C" void kernel_launch(void* const* d_in, const int* in_sizes, int n_in,
                              void* d_out, int out_size, void* d_ws, size_t ws_size,
                              hipStream_t stream) {
  const float* embedding = (const float*)d_in[0];
  const float* scope_enc = (const float*)d_in[1];
  const float* spec_enc  = (const float*)d_in[2];
  const float* W_init    = (const float*)d_in[3];
  const float* b_init    = (const float*)d_in[4];
  const float* W_ih      = (const float*)d_in[5];
  const float* W_hh      = (const float*)d_in[6];
  const float* b_ih      = (const float*)d_in[7];
  const float* b_hh      = (const float*)d_in[8];
  const float* W_out     = (const float*)d_in[9];
  const float* b_out     = (const float*)d_in[10];
  const float* Wd        = (const float*)d_in[11];
  const float* We        = (const float*)d_in[12];
  const float* v_attn    = (const float*)d_in[13];
  const int* tokens_in   = (const int*)d_in[14];
  const int* tokens_out  = (const int*)d_in[15];
  const int* pointer_ids = (const int*)d_in[16];
  const int* ptr_targets = (const int*)d_in[17];
  const unsigned char* pointer_mask = (const unsigned char*)d_in[18];
  float* out = (float*)d_out;

  // f32 workspace region
  float* ws     = (float*)d_ws;
  float* h0f    = ws;                              // 32*512
  float* xprojf = h0f + (long)BB * HH;             // 8192*1536
  float* hdf    = xprojf + (long)BB * TT * 3 * HH; // 8192*512
  float* okf    = hdf + (long)BB * TT * HH;        // 1024*512
  // bf16 workspace region
  __bf16* bw       = (__bf16*)(okf + (long)BB * SS * HH);
  __bf16* obf      = bw;                              // 8192*512
  __bf16* Wih_bf   = obf + (long)BB * TT * HH;        // 1536*1024
  __bf16* Whh_bf   = Wih_bf + (long)3 * HH * 2 * HH;  // 1536*512
  __bf16* Wout_bf  = Whh_bf + (long)3 * HH * HH;      // 128*512
  __bf16* Wd_bf    = Wout_bf + (long)VV * HH;         // 512*512
  __bf16* We_bf    = Wd_bf + (long)HH * HH;           // 512*512
  __bf16* Wi_bf    = We_bf + (long)HH * HH;           // 512*512 (W_init)
  __bf16* emb_bf   = Wi_bf + (long)HH * EE;           // 128*512
  __bf16* scope_bf = emb_bf + (long)VV * HH;          // 1024*512
  __bf16* spec_bf  = scope_bf + (long)BB * SS * HH;   // 32*512
  __bf16* hping    = spec_bf + (long)BB * EE;         // 32*512
  __bf16* hpong    = hping + (long)BB * HH;           // 32*512
  unsigned* syncw  = (unsigned*)(hpong + (long)BB * HH);  // barrier cnt/gen

  zero_out_kernel<<<1, 32, 0, stream>>>(out, BB);
  zero_out_kernel<<<1, 32, 0, stream>>>((float*)syncw, 8);  // cnt=gen=0

  // One-time f32 -> bf16 conversions (weights + GEMM-feeding activations)
  launch_cvt(W_ih, Wih_bf, (long)3 * HH * 2 * HH, stream);
  launch_cvt(W_hh, Whh_bf, (long)3 * HH * HH, stream);
  launch_cvt(W_out, Wout_bf, (long)VV * HH, stream);
  launch_cvt(Wd, Wd_bf, (long)HH * HH, stream);
  launch_cvt(We, We_bf, (long)HH * HH, stream);
  launch_cvt(W_init, Wi_bf, (long)HH * EE, stream);
  launch_cvt(embedding, emb_bf, (long)VV * HH, stream);
  launch_cvt(scope_enc, scope_bf, (long)BB * SS * HH, stream);
  launch_cvt(spec_enc, spec_bf, (long)BB * EE, stream);

  // h0 = spec_enc @ W_init^T + b_init : 2 x 8 (16x64) tiles = 16 waves
  gemm_bb4<<<2, 256, 0, stream>>>(spec_bf, Wi_bf, b_init, h0f, BB, HH, EE);
  // x_proj : 512 x 24 (16x64) tiles = 12288 waves
  xproj_bf4<<<1536, 256, 0, stream>>>(emb_bf, scope_bf, tokens_in, pointer_ids,
                                      pointer_mask, Wih_bf, b_ih, xprojf);
  // recurrence: 64 persistent single-wave blocks, LDS-resident weights
  gru_par<<<64, 32, 0, stream>>>(xprojf, h0f, Whh_bf, b_hh, obf, hping, hpong,
                                 syncw + 0, syncw + 1);
  // token log-likelihood (async-staged LDS weights)
  tok_ll<<<512, 256, 0, stream>>>(obf, Wout_bf, b_out, tokens_out, out);
  // hd = o @ Wd^T : 512 x 8 tiles = 4096 waves
  gemm_bb4<<<512, 256, 0, stream>>>(obf, Wd_bf, nullptr, hdf, BB * TT, HH, HH);
  // ok = scope_enc @ We^T : 64 x 8 tiles = 512 waves
  gemm_bb4<<<64, 256, 0, stream>>>(scope_bf, We_bf, nullptr, okf, BB * SS, HH, HH);
  // additive attention + pointer log-likelihood
  attn_ll<<<BB * TT, 256, 0, stream>>>(hdf, okf, v_attn, ptr_targets,
                                       pointer_mask, out);
}
